// SelfAttentionHead_73589969649855
// MI455X (gfx1250) — matrix operand order repaired and checked by reading, not compiled, not run on hardware
//
#include <hip/hip_runtime.h>

// ---------------------------------------------------------------------------
// Types for CDNA5 WMMA (wave32): D(16x16 f32) = A(16x32 f16) * B(32x16 f16) + C
// ---------------------------------------------------------------------------
typedef __attribute__((ext_vector_type(16))) _Float16 v16h;
typedef __attribute__((ext_vector_type(8)))  _Float16 v8h;
typedef __attribute__((ext_vector_type(8)))  float    v8f;

union V16 { v16h v; v8h h[2]; };

#define WMMA_F16(A, B, C) \
    __builtin_amdgcn_wmma_f32_16x16x32_f16(false, (A), false, (B), (short)0, (C), false, false)

static constexpr int Tn  = 2048;   // sequence length
static constexpr int Cc  = 1024;   // embed dim
static constexpr int Hh  = 64;     // head size
static constexpr int Bb  = 8;      // batch

// ---------------------------------------------------------------------------
// Kernel 1: convert + transpose weights: WT[y][h][c] = W_y[c][h] in f16
// ---------------------------------------------------------------------------
__global__ void wt_prep_kernel(const float* __restrict__ Wq,
                               const float* __restrict__ Wk,
                               const float* __restrict__ Wv,
                               _Float16* __restrict__ WT) {
    const int y = blockIdx.y;
    const float* W = (y == 0) ? Wq : ((y == 1) ? Wk : Wv);
    const int e = blockIdx.x * 256 + threadIdx.x;   // e = c*64 + h (coalesced read)
    if (e < Cc * Hh) {
        const int c = e >> 6;
        const int h = e & 63;
        WT[(size_t)y * (Cc * Hh) + h * Cc + c] = (_Float16)W[e];
    }
}

// ---------------------------------------------------------------------------
// Kernel 2: FUSED q/k/v projections. One wave -> one 16-token M-tile, all
// three 64-feature outputs. x (64 MB) is streamed from HBM exactly ONCE.
// Single scalar loop-carried B pointer; q/k/v + N-tile selected via constant
// element offsets that fold into the 24-bit signed instruction offset.
// launch_bounds(...,1): allow full VGPR file, no accumulator spills.
// ---------------------------------------------------------------------------
__global__ void __launch_bounds__(256, 1)
proj_kernel(const float* __restrict__ x,
            const _Float16* __restrict__ WT,
            _Float16* __restrict__ qs,
            _Float16* __restrict__ ks,
            _Float16* __restrict__ vt) {
    const int lane  = threadIdx.x & 31;
    const int half  = lane >> 4;
    const int lm    = lane & 15;
    const int wave  = threadIdx.x >> 5;
    const int mtile = blockIdx.x * 8 + wave;       // 0..1023 (16384/16 tiles)

    const int row   = mtile * 16 + lm;             // token row this lane loads for A
    const float* xp = x + (size_t)row * Cc + half * 16;

    // scalar loop-carried base: row lm of WT(y=0); +32 halves per K-step
    const _Float16* wb = WT + (size_t)lm * Cc + half * 16;

    v8f accq[4], acck[4], accv[4];
    #pragma unroll
    for (int n = 0; n < 4; ++n)
        #pragma unroll
        for (int r = 0; r < 8; ++r) { accq[n][r] = 0.0f; acck[n][r] = 0.0f; accv[n][r] = 0.0f; }

    for (int kk = 0; kk < Cc / 32; ++kk) {
        // ---- issue ALL loads first; pin them ahead of compute ----
        const float4 f0 = *(const float4*)(xp);
        const float4 f1 = *(const float4*)(xp + 4);
        const float4 f2 = *(const float4*)(xp + 8);
        const float4 f3 = *(const float4*)(xp + 12);
        v16h bq[4], bk[4], bv[4];
        #pragma unroll
        for (int n = 0; n < 4; ++n) bq[n] = *(const v16h*)(wb + n * (16 * Cc));
        #pragma unroll
        for (int n = 0; n < 4; ++n) bk[n] = *(const v16h*)(wb + 1 * Cc * Hh + n * (16 * Cc));
        #pragma unroll
        for (int n = 0; n < 4; ++n) bv[n] = *(const v16h*)(wb + 2 * Cc * Hh + n * (16 * Cc));
        asm volatile("" ::: "memory");   // keep loads clustered before WMMAs

        // ---- A fragment f32 -> f16 (v_cvt_pk_f16_f32 pairs) ----
        V16 a;
        a.v[0]=(_Float16)f0.x; a.v[1]=(_Float16)f0.y; a.v[2]=(_Float16)f0.z; a.v[3]=(_Float16)f0.w;
        a.v[4]=(_Float16)f1.x; a.v[5]=(_Float16)f1.y; a.v[6]=(_Float16)f1.z; a.v[7]=(_Float16)f1.w;
        a.v[8]=(_Float16)f2.x; a.v[9]=(_Float16)f2.y; a.v[10]=(_Float16)f2.z; a.v[11]=(_Float16)f2.w;
        a.v[12]=(_Float16)f3.x; a.v[13]=(_Float16)f3.y; a.v[14]=(_Float16)f3.z; a.v[15]=(_Float16)f3.w;

        // ---- 12 WMMAs sharing one A fragment ----
        #pragma unroll
        for (int n = 0; n < 4; ++n) accq[n] = WMMA_F16(a.v, bq[n], accq[n]);
        #pragma unroll
        for (int n = 0; n < 4; ++n) acck[n] = WMMA_F16(a.v, bk[n], acck[n]);
        #pragma unroll
        for (int n = 0; n < 4; ++n) accv[n] = WMMA_F16(a.v, bv[n], accv[n]);

        xp += 32;
        wb += 32;
    }

    // ---- store tiles (C/D layout: VGPR r -> row r + 8*half, col = n*16 + lm)
    #pragma unroll
    for (int n = 0; n < 4; ++n)
        #pragma unroll
        for (int r = 0; r < 8; ++r) {
            const int tok = mtile * 16 + r + 8 * half;
            qs[(size_t)tok * Hh + n * 16 + lm] = (_Float16)(accq[n][r] * 0.125f);
        }
    #pragma unroll
    for (int n = 0; n < 4; ++n)
        #pragma unroll
        for (int r = 0; r < 8; ++r) {
            const int tok = mtile * 16 + r + 8 * half;
            ks[(size_t)tok * Hh + n * 16 + lm] = (_Float16)acck[n][r];
        }
    #pragma unroll
    for (int n = 0; n < 4; ++n)
        #pragma unroll
        for (int r = 0; r < 8; ++r) {
            const int tok = mtile * 16 + r + 8 * half;
            const int bb  = tok >> 11;
            const int tt  = tok & (Tn - 1);
            vt[((size_t)(bb * Hh + n * 16 + lm)) * Tn + tt] = (_Float16)accv[n][r];
        }
}

// ---------------------------------------------------------------------------
// Kernel 3: causal flash attention. One wave per 16-row query tile,
// online softmax over 32-wide key blocks. P transposed via per-wave LDS slice.
// ---------------------------------------------------------------------------
__global__ void __launch_bounds__(128, 1)
attn_kernel(const _Float16* __restrict__ qs,
            const _Float16* __restrict__ ks,
            const _Float16* __restrict__ vt,
            float* __restrict__ out) {
    __shared__ _Float16 lP[4][16 * 32];            // per-wave P tile (row-major 16x32)

    const int lane = threadIdx.x & 31;
    const int half = lane >> 4;
    const int lm   = lane & 15;
    const int wave = threadIdx.x >> 5;
    const int qt   = blockIdx.x * 4 + wave;        // query tile (16 rows), 0..127
    const int b    = blockIdx.y;
    _Float16* P    = lP[wave];

    // ---- Q A-fragments (K = head dim, split 0..31 / 32..63), loaded once
    const _Float16* qrow = qs + (size_t)(b * Tn + qt * 16 + lm) * Hh;
    V16 qa0, qa1;
    qa0.h[0] = *(const v8h*)(qrow + half * 16);
    qa0.h[1] = *(const v8h*)(qrow + 8 + half * 16);
    qa1.h[0] = *(const v8h*)(qrow + 32 + half * 16);
    qa1.h[1] = *(const v8h*)(qrow + 40 + half * 16);

    float m[8], l[8];
    v8f acc[4];
    #pragma unroll
    for (int r = 0; r < 8; ++r) { m[r] = -1.0e30f; l[r] = 0.0f; }
    #pragma unroll
    for (int n = 0; n < 4; ++n)
        #pragma unroll
        for (int r = 0; r < 8; ++r) acc[n][r] = 0.0f;

    const int jlast = (qt * 16 + 15) >> 5;

    // scalar loop-carried pointers (address-space inference friendly)
    const _Float16* kb = ks + (size_t)(b * Tn) * Hh;               // +32*Hh per j
    const _Float16* vb = vt + ((size_t)b * Hh + lm) * Tn + half * 16;  // +32 per j

    for (int j = 0; j <= jlast; ++j) {
        if (j < jlast) {   // prefetch next K block -> global_prefetch_b8
            __builtin_prefetch(kb + (size_t)32 * Hh + lane * 64, 0, 1);
        }

        // ---- issue all global loads up front: K frags then V frags ----
        const v16h kb00 = *(const v16h*)(kb + (size_t)(lm)      * Hh +      half * 16);
        const v16h kb01 = *(const v16h*)(kb + (size_t)(lm)      * Hh + 32 + half * 16);
        const v16h kb10 = *(const v16h*)(kb + (size_t)(16 + lm) * Hh +      half * 16);
        const v16h kb11 = *(const v16h*)(kb + (size_t)(16 + lm) * Hh + 32 + half * 16);
        v16h vf[4];
        #pragma unroll
        for (int n = 0; n < 4; ++n)
            vf[n] = *(const v16h*)(vb + n * (16 * Tn));  // in flight over softmax

        // ---- S = Q K^T : two 16-col subtiles, head dim split in two 32-chunks
        v8f s0, s1;
        #pragma unroll
        for (int r = 0; r < 8; ++r) { s0[r] = 0.0f; s1[r] = 0.0f; }
        s0 = WMMA_F16(qa0.v, kb00, s0);
        s0 = WMMA_F16(qa1.v, kb01, s0);
        s1 = WMMA_F16(qa0.v, kb10, s1);
        s1 = WMMA_F16(qa1.v, kb11, s1);

        // ---- causal mask (only the diagonal block(s) actually mask)
        const bool diag = (j * 32 + 31) > (qt * 16);
        float mj[8];
        #pragma unroll
        for (int r = 0; r < 8; ++r) {
            float v0 = s0[r], v1 = s1[r];
            if (diag) {
                const int rowg = qt * 16 + r + 8 * half;
                if (j * 32 + lm      > rowg) v0 = -1.0e30f;
                if (j * 32 + 16 + lm > rowg) v1 = -1.0e30f;
            }
            s0[r] = v0; s1[r] = v1;
            mj[r] = fmaxf(v0, v1);
        }
        // row-max across the 16 lanes holding each row
        #pragma unroll
        for (int r = 0; r < 8; ++r) {
            float v = mj[r];
            v = fmaxf(v, __shfl_xor(v, 1));
            v = fmaxf(v, __shfl_xor(v, 2));
            v = fmaxf(v, __shfl_xor(v, 4));
            v = fmaxf(v, __shfl_xor(v, 8));
            mj[r] = v;
        }

        // ---- online softmax update
        float alpha[8];
        #pragma unroll
        for (int r = 0; r < 8; ++r) {
            const float mn = fmaxf(m[r], mj[r]);
            alpha[r] = __expf(m[r] - mn);
            m[r] = mn;
            const float p0 = __expf(s0[r] - mn);
            const float p1 = __expf(s1[r] - mn);
            s0[r] = p0; s1[r] = p1;
            float ls = p0 + p1;
            ls += __shfl_xor(ls, 1);
            ls += __shfl_xor(ls, 2);
            ls += __shfl_xor(ls, 4);
            ls += __shfl_xor(ls, 8);
            l[r] = l[r] * alpha[r] + ls;
        }
        #pragma unroll
        for (int n = 0; n < 4; ++n)
            #pragma unroll
            for (int r = 0; r < 8; ++r) acc[n][r] *= alpha[r];

        // ---- transpose P (C-layout -> A-layout) through LDS, f32 -> f16
        #pragma unroll
        for (int r = 0; r < 8; ++r) {
            const int rr = r + 8 * half;
            P[rr * 32 + lm]      = (_Float16)s0[r];
            P[rr * 32 + 16 + lm] = (_Float16)s1[r];
        }
        asm volatile("" ::: "memory");   // keep ds stores before ds loads
        V16 pa;
        pa.h[0] = *(const v8h*)(P + lm * 32 + half * 16);
        pa.h[1] = *(const v8h*)(P + lm * 32 + 8 + half * 16);
        asm volatile("" ::: "memory");

        // ---- O += P * V  (V^T stored: contiguous B-fragments, already loaded)
        #pragma unroll
        for (int n = 0; n < 4; ++n) acc[n] = WMMA_F16(pa.v, vf[n], acc[n]);

        kb += (size_t)32 * Hh;
        vb += 32;
    }

    // ---- normalize and store f32 output
    float inv[8];
    #pragma unroll
    for (int r = 0; r < 8; ++r) inv[r] = 1.0f / l[r];
    #pragma unroll
    for (int n = 0; n < 4; ++n)
        #pragma unroll
        for (int r = 0; r < 8; ++r) {
            const int tok = b * Tn + qt * 16 + r + 8 * half;
            out[(size_t)tok * Hh + n * 16 + lm] = acc[n][r] * inv[r];
        }
}

// ---------------------------------------------------------------------------
// Launch: wt_prep -> proj (fused qkv) -> attn (same stream, sequential)
// ---------------------------------------------------------------------------
extern "C" void kernel_launch(void* const* d_in, const int* in_sizes, int n_in,
                              void* d_out, int out_size, void* d_ws, size_t ws_size,
                              hipStream_t stream) {
    const float* x  = (const float*)d_in[0];
    const float* Wq = (const float*)d_in[1];
    const float* Wk = (const float*)d_in[2];
    const float* Wv = (const float*)d_in[3];
    float* out = (float*)d_out;

    // workspace layout (f16): WT[3][64][1024] | qs[16384][64] | ks[16384][64] | vt[8][64][2048]
    _Float16* WT  = (_Float16*)d_ws;
    _Float16* qsb = WT + (size_t)3 * Cc * Hh;
    _Float16* ksb = qsb + (size_t)Bb * Tn * Hh;
    _Float16* vtb = ksb + (size_t)Bb * Tn * Hh;

    wt_prep_kernel<<<dim3(256, 3), 256, 0, stream>>>(Wq, Wk, Wv, WT);
    proj_kernel<<<128, 256, 0, stream>>>(x, WT, qsb, ksb, vtb);
    attn_kernel<<<dim3(32, 8), 128, 0, stream>>>(qsb, ksb, vtb, out);
}
